// InteractionHead_7421703488011
// MI455X (gfx1250) — matrix operand order repaired and checked by reading, not compiled, not run on hardware
//
#include <hip/hip_runtime.h>
#include <hip/hip_bf16.h>

typedef __attribute__((ext_vector_type(16))) _Float16 v16h;
typedef __attribute__((ext_vector_type(8)))  _Float16 v8h;
typedef __attribute__((ext_vector_type(2)))  _Float16 v2h;
typedef __attribute__((ext_vector_type(8)))  float    v8f;
typedef __attribute__((ext_vector_type(4)))  float    v4f;

#define BM 128
#define BN 256
#define BK 32
#define LDSA 40   // 32 + 8 halfs pad -> 80B row stride, conflict-free b128 loads
#define LDSB 40

__device__ __forceinline__ v16h cat8(v8h lo, v8h hi) {
  return __builtin_shufflevector(lo, hi, 0,1,2,3,4,5,6,7,8,9,10,11,12,13,14,15);
}

// EPI = 0 : D = relu(A@B + bias)                              -> f16 store
// EPI = 1 : D = ps[m]*obj_mask[lab[m],n]*sigmoid(A@B + bias)  -> f32 store
template<int EPI, typename TA>
__global__ __launch_bounds__(256)
void gemm_wmma_kernel(const TA*    __restrict__ A, int lda,     // M x K row major
                      const float* __restrict__ B, int ldb,     // K x N row major
                      const float* __restrict__ bias,           // N
                      void*        __restrict__ Cout, int ldc,  // M x N
                      int Ndim, int Kdim,
                      const float* __restrict__ ps,
                      const int*   __restrict__ lab,
                      const float* __restrict__ obj_mask)
{
  __shared__ _Float16 As[2][BM * LDSA];   // 2 x 10240 B
  __shared__ _Float16 Bs[2][BN * LDSB];   // 2 x 20480 B (transposed: Bs[n][k])

  const int tid   = threadIdx.x;
  const int lane  = tid & 31;
  const int wave  = tid >> 5;
  const int wm    = wave >> 2;      // 0..1  -> 64-row slab
  const int wn    = wave & 3;       // 0..3  -> 64-col slab
  const int lhalf = lane >> 4;
  const int l15   = lane & 15;

  const int bm = blockIdx.y * BM;
  const int bn = blockIdx.x * BN;

  // A staging: thread -> (row 0..127, 16-wide k chunk)
  const int ar = tid >> 1;
  const int ak = (tid & 1) * 16;
  // B staging: thread -> (k pair 0..15, 16-wide n chunk)
  const int kp = tid >> 4;            // k = 2*kp, 2*kp+1
  const int nb = (tid & 15) * 16;     // 0..240

  v8f zero = {0.f,0.f,0.f,0.f,0.f,0.f,0.f,0.f};
  v8f acc[4][4];
  #pragma unroll
  for (int i = 0; i < 4; ++i)
    #pragma unroll
    for (int j = 0; j < 4; ++j)
      acc[i][j] = zero;

  // ---------------- prologue: stage K-tile 0 into buffer 0 ----------------
  {
    const TA* aptr = A + (size_t)(bm + ar) * lda + ak;
    if constexpr (sizeof(TA) == 4) {
      v4f f0 = *(const v4f*)(aptr + 0);
      v4f f1 = *(const v4f*)(aptr + 4);
      v4f f2 = *(const v4f*)(aptr + 8);
      v4f f3 = *(const v4f*)(aptr + 12);
      v8h h0, h1;
      #pragma unroll
      for (int j = 0; j < 4; ++j) {
        h0[j] = (_Float16)f0[j]; h0[4+j] = (_Float16)f1[j];
        h1[j] = (_Float16)f2[j]; h1[4+j] = (_Float16)f3[j];
      }
      *(v8h*)&As[0][ar * LDSA + ak]     = h0;
      *(v8h*)&As[0][ar * LDSA + ak + 8] = h1;
    } else {
      const _Float16* ap = (const _Float16*)aptr;
      *(v8h*)&As[0][ar * LDSA + ak]     = *(const v8h*)(ap);
      *(v8h*)&As[0][ar * LDSA + ak + 8] = *(const v8h*)(ap + 8);
    }
    const float* b0 = B + (size_t)(2 * kp) * ldb + bn + nb;
    const float* b1 = b0 + ldb;
    #pragma unroll
    for (int g = 0; g < 4; ++g) {
      const int ncol = bn + nb + g * 4;
      v4f vA, vB;
      if (ncol + 4 <= Ndim) {
        vA = *(const v4f*)(b0 + g * 4);
        vB = *(const v4f*)(b1 + g * 4);
      } else {
        #pragma unroll
        for (int j = 0; j < 4; ++j) {
          vA[j] = (ncol + j < Ndim) ? b0[g * 4 + j] : 0.f;
          vB[j] = (ncol + j < Ndim) ? b1[g * 4 + j] : 0.f;
        }
      }
      #pragma unroll
      for (int j = 0; j < 4; ++j) {
        v2h pr = { (_Float16)vA[j], (_Float16)vB[j] };
        *(v2h*)&Bs[0][(nb + g * 4 + j) * LDSB + 2 * kp] = pr;
      }
    }
  }

  // ---------------- main loop: double-buffered, register-prefetched --------
  const int iters = Kdim / BK;
  for (int it = 0; it < iters; ++it) {
    const int cur = it & 1;
    __syncthreads();
    const bool more = (it + 1 < iters);

    // issue next K-tile global loads early (wait lands after the WMMA block)
    v4f aF0, aF1, aF2, aF3;
    v8h aH0, aH1;
    v4f bA[4], bB[4];
    if (more) {
      const int kn = (it + 1) * BK;
      const TA* aptr = A + (size_t)(bm + ar) * lda + kn + ak;
      if constexpr (sizeof(TA) == 4) {
        aF0 = *(const v4f*)(aptr + 0);
        aF1 = *(const v4f*)(aptr + 4);
        aF2 = *(const v4f*)(aptr + 8);
        aF3 = *(const v4f*)(aptr + 12);
      } else {
        const _Float16* ap = (const _Float16*)aptr;
        aH0 = *(const v8h*)(ap);
        aH1 = *(const v8h*)(ap + 8);
      }
      const float* b0 = B + (size_t)(kn + 2 * kp) * ldb + bn + nb;
      const float* b1 = b0 + ldb;
      #pragma unroll
      for (int g = 0; g < 4; ++g) {
        const int ncol = bn + nb + g * 4;
        if (ncol + 4 <= Ndim) {
          bA[g] = *(const v4f*)(b0 + g * 4);
          bB[g] = *(const v4f*)(b1 + g * 4);
        } else {
          #pragma unroll
          for (int j = 0; j < 4; ++j) {
            bA[g][j] = (ncol + j < Ndim) ? b0[g * 4 + j] : 0.f;
            bB[g][j] = (ncol + j < Ndim) ? b1[g * 4 + j] : 0.f;
          }
        }
      }
    }

    // compute on buffer `cur` (ISA 7.12.2 fragment layouts)
    v16h bf[4];
    #pragma unroll
    for (int nt = 0; nt < 4; ++nt) {
      const int col = wn * 64 + nt * 16 + l15;
      const _Float16* p = &Bs[cur][col * LDSB + lhalf * 8]; // K = lhalf*8+0..7, 16+lhalf*8+0..7
      bf[nt] = cat8(*(const v8h*)(p), *(const v8h*)(p + 16));
    }
    #pragma unroll
    for (int mt = 0; mt < 4; ++mt) {
      const int row = wm * 64 + mt * 16 + l15;
      const _Float16* p = &As[cur][row * LDSA + lhalf * 8];
      v16h a = cat8(*(const v8h*)(p),        // K = lhalf*8 + 0..7
                    *(const v8h*)(p + 16));  // K = 16 + lhalf*8 + 0..7
      #pragma unroll
      for (int nt = 0; nt < 4; ++nt)
        acc[mt][nt] = __builtin_amdgcn_wmma_f32_16x16x32_f16(
            false, a, false, bf[nt], (short)0, acc[mt][nt], false, false);
    }

    // convert + store next tile into the other buffer
    if (more) {
      const int nxt = cur ^ 1;
      if constexpr (sizeof(TA) == 4) {
        v8h h0, h1;
        #pragma unroll
        for (int j = 0; j < 4; ++j) {
          h0[j] = (_Float16)aF0[j]; h0[4+j] = (_Float16)aF1[j];
          h1[j] = (_Float16)aF2[j]; h1[4+j] = (_Float16)aF3[j];
        }
        *(v8h*)&As[nxt][ar * LDSA + ak]     = h0;
        *(v8h*)&As[nxt][ar * LDSA + ak + 8] = h1;
      } else {
        *(v8h*)&As[nxt][ar * LDSA + ak]     = aH0;
        *(v8h*)&As[nxt][ar * LDSA + ak + 8] = aH1;
      }
      #pragma unroll
      for (int g = 0; g < 4; ++g)
        #pragma unroll
        for (int j = 0; j < 4; ++j) {
          v2h pr = { (_Float16)bA[g][j], (_Float16)bB[g][j] };
          *(v2h*)&Bs[nxt][(nb + g * 4 + j) * LDSB + 2 * kp] = pr;
        }
    }
  }

  // ---------------- epilogue ----------------
  #pragma unroll
  for (int mt = 0; mt < 4; ++mt) {
    #pragma unroll
    for (int nt = 0; nt < 4; ++nt) {
      const int n = bn + wn * 64 + nt * 16 + l15;
      const float bv = (n < Ndim) ? bias[n] : 0.f;
      #pragma unroll
      for (int v = 0; v < 8; ++v) {
        const int m = bm + wm * 64 + mt * 16 + v + 8 * lhalf;
        float val = acc[mt][nt][v] + bv;
        if constexpr (EPI == 0) {
          val = val > 0.f ? val : 0.f;
          ((_Float16*)Cout)[(size_t)m * ldc + n] = (_Float16)val;
        } else {
          if (n < Ndim) {
            const float sg = 1.0f / (1.0f + __expf(-val));
            ((float*)Cout)[(size_t)m * ldc + n] =
                ps[m] * obj_mask[lab[m] * Ndim + n] * sg;
          }
        }
      }
    }
  }
}

__global__ void pair_prep_kernel(const int* __restrict__ paired_idx,
                                 const float* __restrict__ scores,
                                 const int* __restrict__ labels,
                                 float* __restrict__ ps,
                                 int* __restrict__ lab,
                                 int Mrows)
{
  const int m = blockIdx.x * blockDim.x + threadIdx.x;
  if (m < Mrows) {
    const int hi = paired_idx[2 * m + 0];
    const int oi = paired_idx[2 * m + 1];
    ps[m]  = scores[hi] * scores[oi];
    lab[m] = labels[oi];
  }
}

extern "C" void kernel_launch(void* const* d_in, const int* in_sizes, int n_in,
                              void* d_out, int out_size, void* d_ws, size_t ws_size,
                              hipStream_t stream) {
  const float* features   = (const float*)d_in[0];   // 16384 x 12544
  const float* W1         = (const float*)d_in[1];   // 12544 x 1024
  const float* b1         = (const float*)d_in[2];   // 1024
  const float* W2         = (const float*)d_in[3];   // 1024 x 1024
  const float* b2         = (const float*)d_in[4];   // 1024
  const float* Wc         = (const float*)d_in[5];   // 1024 x 600
  const float* bc         = (const float*)d_in[6];   // 600
  const float* scores     = (const float*)d_in[7];   // 128
  const float* obj_mask   = (const float*)d_in[8];   // 80 x 600
  const int*   labels     = (const int*)d_in[9];     // 128
  const int*   paired_idx = (const int*)d_in[10];    // 16384 x 2
  float* out = (float*)d_out;                        // 16384 x 600 f32

  const int Mr = 16384, DIN = 12544, DREP = 1024, KC = 600;

  // workspace: h1 (f16 M x 1024) | h2 (f16 M x 1024) | ps (f32 M) | lab (i32 M)
  _Float16* h1 = (_Float16*)d_ws;
  _Float16* h2 = h1 + (size_t)Mr * DREP;
  float*    ps = (float*)(h2 + (size_t)Mr * DREP);
  int*      lab = (int*)(ps + Mr);

  pair_prep_kernel<<<Mr / 256, 256, 0, stream>>>(paired_idx, scores, labels, ps, lab, Mr);

  dim3 blk(256);
  // GEMM1: h1 = relu(features @ W1 + b1)
  gemm_wmma_kernel<0, float><<<dim3(DREP / BN, Mr / BM), blk, 0, stream>>>(
      features, DIN, W1, DREP, b1, (void*)h1, DREP, DREP, DIN,
      nullptr, nullptr, nullptr);
  // GEMM2: h2 = relu(h1 @ W2 + b2)
  gemm_wmma_kernel<0, _Float16><<<dim3(DREP / BN, Mr / BM), blk, 0, stream>>>(
      h1, DREP, W2, DREP, b2, (void*)h2, DREP, DREP, DREP,
      nullptr, nullptr, nullptr);
  // GEMM3 + fused epilogue: out = ps * obj_mask[lab] * sigmoid(h2 @ Wc + bc)
  gemm_wmma_kernel<1, _Float16><<<dim3((KC + BN - 1) / BN, Mr / BM), blk, 0, stream>>>(
      h2, DREP, Wc, KC, bc, (void*)out, KC, KC, DREP,
      ps, lab, obj_mask);
}
